// TransformerBlock_75230647157345
// MI455X (gfx1250) — compile-verified, matrix operand order checked by reading, not run on hardware
//
#include <hip/hip_runtime.h>
#include <math.h>

#define D_MODEL 1024
#define N_HEADS 16
#define D_HEAD  64
#define D_FF    4096
#define BATCH   2
#define SEQ     2048
#define BT      (BATCH*SEQ)

typedef __attribute__((ext_vector_type(16))) __bf16       v16bf;
typedef __attribute__((ext_vector_type(8)))  float        v8f;
typedef __attribute__((ext_vector_type(4)))  unsigned int u32x4;
typedef __attribute__((ext_vector_type(2)))  unsigned int u32x2;
typedef __attribute__((ext_vector_type(4)))  float        f32x4;

// async-DMA builtin takes int4 pointers: vector_size(16) int, addrspace 1 (global) / 3 (LDS)
typedef int v4i_t __attribute__((vector_size(16)));
typedef __attribute__((address_space(1))) v4i_t as1_v4i;
typedef __attribute__((address_space(3))) v4i_t as3_v4i;

static __device__ __forceinline__ unsigned short f2bf(float f) {
  union { float f; unsigned u; } c; c.f = f;
  unsigned u = c.u + 0x7FFFu + ((c.u >> 16) & 1u);   // round-to-nearest-even
  return (unsigned short)(u >> 16);
}

union FragU { v16bf v; u32x4 q[2]; };

// 16B global -> LDS copy: async DMA (ASYNCcnt) when available, else via VGPRs.
static __device__ __forceinline__ void ld128_to_lds(const unsigned short* gp, unsigned short* lp) {
#if __has_builtin(__builtin_amdgcn_global_load_async_to_lds_b128)
  __builtin_amdgcn_global_load_async_to_lds_b128(
      (as1_v4i*)(__UINTPTR_TYPE__)gp,
      (as3_v4i*)(unsigned)(__UINTPTR_TYPE__)lp, 0, 0);
#else
  *(u32x4*)lp = *(const u32x4*)gp;
#endif
}
static __device__ __forceinline__ void wait_async_le4() {
#if __has_builtin(__builtin_amdgcn_s_wait_asynccnt)
  __builtin_amdgcn_s_wait_asynccnt(4);
#else
  asm volatile("s_wait_asynccnt 0x4" ::: "memory");
#endif
}
static __device__ __forceinline__ void wait_async_le0() {
#if __has_builtin(__builtin_amdgcn_s_wait_asynccnt)
  __builtin_amdgcn_s_wait_asynccnt(0);
#else
  asm volatile("s_wait_asynccnt 0x0" ::: "memory");
#endif
}

// butterfly reductions across 16-lane groups via DPP16 (VALU, no LDS round-trips)
// xor1 = quad_perm(1,0,3,2)=0xB1 ; xor2 = quad_perm(2,3,0,1)=0x4E ;
// xor7-effective = row_half_mirror(0x141) on quad-uniform data ; xor15-effective = row_mirror(0x140)
static __device__ __forceinline__ float red16_max(float x) {
#if __has_builtin(__builtin_amdgcn_update_dpp)
  int v;
  v = __builtin_amdgcn_update_dpp(__float_as_int(x), __float_as_int(x), 0xB1, 0xF, 0xF, true);
  x = fmaxf(x, __int_as_float(v));
  v = __builtin_amdgcn_update_dpp(__float_as_int(x), __float_as_int(x), 0x4E, 0xF, 0xF, true);
  x = fmaxf(x, __int_as_float(v));
  v = __builtin_amdgcn_update_dpp(__float_as_int(x), __float_as_int(x), 0x141, 0xF, 0xF, true);
  x = fmaxf(x, __int_as_float(v));
  v = __builtin_amdgcn_update_dpp(__float_as_int(x), __float_as_int(x), 0x140, 0xF, 0xF, true);
  x = fmaxf(x, __int_as_float(v));
#else
  for (int mk = 1; mk < 16; mk <<= 1) x = fmaxf(x, __shfl_xor(x, mk, 32));
#endif
  return x;
}
static __device__ __forceinline__ float red16_sum(float x) {
#if __has_builtin(__builtin_amdgcn_update_dpp)
  int v;
  v = __builtin_amdgcn_update_dpp(__float_as_int(x), __float_as_int(x), 0xB1, 0xF, 0xF, true);
  x += __int_as_float(v);
  v = __builtin_amdgcn_update_dpp(__float_as_int(x), __float_as_int(x), 0x4E, 0xF, 0xF, true);
  x += __int_as_float(v);
  v = __builtin_amdgcn_update_dpp(__float_as_int(x), __float_as_int(x), 0x141, 0xF, 0xF, true);
  x += __int_as_float(v);
  v = __builtin_amdgcn_update_dpp(__float_as_int(x), __float_as_int(x), 0x140, 0xF, 0xF, true);
  x += __int_as_float(v);
#else
  for (int mk = 1; mk < 16; mk <<= 1) x += __shfl_xor(x, mk, 32);
#endif
  return x;
}

// ---------------------------------------------------------------- fp32 [K,N] -> bf16 [N,K] (transpose+convert)
__global__ __launch_bounds__(256) void cvt_transpose_bf16(const float* __restrict__ W,
                                                          unsigned short* __restrict__ Wt,
                                                          int K, int N) {
  __shared__ unsigned short tile[32][33];
  const int k0 = blockIdx.y * 32;
  const int n0 = blockIdx.x * 32;
  const int t  = threadIdx.x;
  const int r  = t >> 3;          // 0..31
  const int c4 = (t & 7) * 4;     // 0..28
  f32x4 v = *(const f32x4*)(W + (size_t)(k0 + r) * N + n0 + c4);
  tile[c4 + 0][r] = f2bf(v.x);
  tile[c4 + 1][r] = f2bf(v.y);
  tile[c4 + 2][r] = f2bf(v.z);
  tile[c4 + 3][r] = f2bf(v.w);
  __syncthreads();
  unsigned short o[4];
  o[0] = tile[r][c4 + 0]; o[1] = tile[r][c4 + 1];
  o[2] = tile[r][c4 + 2]; o[3] = tile[r][c4 + 3];
  *(u32x2*)(Wt + (size_t)(n0 + r) * K + k0 + c4) = *(const u32x2*)o;
}

// ---------------------------------------------------------------- LayerNorm -> bf16
__global__ __launch_bounds__(256) void layernorm_to_bf16(const float* __restrict__ x,
                                                         const float* __restrict__ gamma,
                                                         const float* __restrict__ beta,
                                                         unsigned short* __restrict__ out) {
  int row = blockIdx.x;
  int t   = threadIdx.x;
  const float* xr = x + (size_t)row * D_MODEL;
  f32x4 v = *(const f32x4*)(xr + t * 4);
  float s  = v.x + v.y + v.z + v.w;
  float s2 = v.x*v.x + v.y*v.y + v.z*v.z + v.w*v.w;
  for (int m = 1; m < 32; m <<= 1) { s += __shfl_xor(s, m, 32); s2 += __shfl_xor(s2, m, 32); }
  __shared__ float rs[8], rq[8];
  int w = t >> 5, lane = t & 31;
  if (lane == 0) { rs[w] = s; rq[w] = s2; }
  __syncthreads();
  s = 0.f; s2 = 0.f;
  for (int i = 0; i < 8; ++i) { s += rs[i]; s2 += rq[i]; }
  const float inv = 1.0f / D_MODEL;
  float mu   = s * inv;
  float var  = s2 * inv - mu * mu;
  float rstd = rsqrtf(var + 1e-5f);
  unsigned short o[4];
  const float* ve = (const float*)&v;
  #pragma unroll
  for (int j = 0; j < 4; ++j) {
    int c = t * 4 + j;
    o[j] = f2bf((ve[j] - mu) * rstd * gamma[c] + beta[c]);
  }
  *(u32x2*)(out + (size_t)row * D_MODEL + t * 4) = *(const u32x2*)o;
}

// ---------------------------------------------------------------- WMMA GEMM (double-buffered, async tiles)
// C[M,N] = epilogue(A[M,K]_bf16 x Bt[N,K]_bf16^T)
// EPI: 0 = store bf16 ; 1 = bias + GELU -> bf16 ; 2 = Res + acc -> f32 ; 3 = Res + acc + bias -> f32
#define BMT 128
#define BNT 128
#define BKT 32
#define LDST 40   // halves per LDS tile row (32 + 8 pad)

// one 128x32 tile: 512 16B chunks, 2 per thread (256 threads) -> 2 async ops/thread
static __device__ __forceinline__ void tile_copy(unsigned short* dst,
                                                 const unsigned short* src, size_t ld,
                                                 int k0, int tid) {
  #pragma unroll
  for (int c = 0; c < 2; ++c) {
    int chunk = tid + c * 256;
    int r  = chunk >> 2;
    int c8 = (chunk & 3) * 8;
    ld128_to_lds(src + (size_t)r * ld + k0 + c8, dst + r * LDST + c8);
  }
}

template <int EPI>
__global__ __launch_bounds__(256) void gemm_bf16_wmma(
    const unsigned short* __restrict__ A,    // [M,K]
    const unsigned short* __restrict__ Bt,   // [N,K] pre-transposed
    void* __restrict__ Cout,
    const float* __restrict__ bias,
    const float* __restrict__ Res,
    int M, int N, int K)
{
  __shared__ unsigned short sA[2][BMT * LDST];
  __shared__ unsigned short sB[2][BNT * LDST];

  const int tid  = threadIdx.x;
  const int wave = tid >> 5;
  const int lane = tid & 31;
  const int wm   = wave >> 1;   // 0..3  (32-row slab)
  const int wn   = wave & 1;    // 0..1  (64-col slab)
  const int hf   = lane >> 4;   // 0/1
  const int l15  = lane & 15;
  const int bm0  = blockIdx.y * BMT;
  const int bn0  = blockIdx.x * BNT;

  const unsigned short* Ab = A  + (size_t)bm0 * K;
  const unsigned short* Bb = Bt + (size_t)bn0 * K;

  const v8f vzero = {0.f,0.f,0.f,0.f,0.f,0.f,0.f,0.f};
  v8f acc[2][4];
  for (int i = 0; i < 2; ++i) for (int j = 0; j < 4; ++j) acc[i][j] = vzero;

  // prologue: issue tile 0
  tile_copy(sA[0], Ab, K, 0, tid);
  tile_copy(sB[0], Bb, K, 0, tid);

  const int nk = K / BKT;
  for (int kt = 0; kt < nk; ++kt) {
    const int cur = kt & 1;
    if (kt + 1 < nk) {  // prefetch next tile into other buffer, retire only current 4 ops
      tile_copy(sA[cur ^ 1], Ab, K, (kt + 1) * BKT, tid);
      tile_copy(sB[cur ^ 1], Bb, K, (kt + 1) * BKT, tid);
      wait_async_le4();
    } else {
      wait_async_le0();
    }
    __syncthreads();

    FragU af[2], bf[4];
    #pragma unroll
    for (int tm = 0; tm < 2; ++tm) {
      const unsigned short* p = sA[cur] + (wm * 32 + tm * 16 + l15) * LDST + hf * 8;
      af[tm].q[0] = *(const u32x4*)p;
      af[tm].q[1] = *(const u32x4*)(p + 16);
    }
    #pragma unroll
    for (int tn = 0; tn < 4; ++tn) {
      const unsigned short* p = sB[cur] + (wn * 64 + tn * 16 + l15) * LDST + hf * 8;
      bf[tn].q[0] = *(const u32x4*)p;
      bf[tn].q[1] = *(const u32x4*)(p + 16);
    }
    #pragma unroll
    for (int tm = 0; tm < 2; ++tm)
      #pragma unroll
      for (int tn = 0; tn < 4; ++tn)
        acc[tm][tn] = __builtin_amdgcn_wmma_f32_16x16x32_bf16(
            false, af[tm].v, false, bf[tn].v, (short)0, acc[tm][tn], false, false);
    __syncthreads();
  }

  #pragma unroll
  for (int tm = 0; tm < 2; ++tm) {
    #pragma unroll
    for (int tn = 0; tn < 4; ++tn) {
      int col = bn0 + wn * 64 + tn * 16 + l15;
      float bv = 0.f;
      if constexpr (EPI == 1 || EPI == 3) bv = bias[col];
      #pragma unroll
      for (int r = 0; r < 8; ++r) {
        int row = bm0 + wm * 32 + tm * 16 + hf * 8 + r;
        size_t idx = (size_t)row * N + col;
        float v = acc[tm][tn][r];
        if constexpr (EPI == 0) {
          ((unsigned short*)Cout)[idx] = f2bf(v);
        } else if constexpr (EPI == 1) {
          v += bv;
          v = 0.5f * v * (1.0f + erff(v * 0.70710678118654752f));
          ((unsigned short*)Cout)[idx] = f2bf(v);
        } else if constexpr (EPI == 2) {
          ((float*)Cout)[idx] = Res[idx] + v;
        } else {
          ((float*)Cout)[idx] = Res[idx] + v + bv;
        }
      }
    }
  }
}

// ---------------------------------------------------------------- flash attention (WMMA)
#define LDK 72   // 64 + 8 pad halves
#define LDV 40   // 32 + 8 pad halves
#define LDP 40

__global__ __launch_bounds__(128) void attn_wmma(
    const unsigned short* __restrict__ qkv,   // [BT, 3*D_MODEL] bf16
    unsigned short* __restrict__ outp)        // [BT, D_MODEL]  bf16
{
  const int bh = blockIdx.y;
  const int b  = bh >> 4;
  const int h  = bh & 15;
  const int q0 = blockIdx.x * 64;

  const int tid  = threadIdx.x;
  const int wave = tid >> 5;
  const int lane = tid & 31;
  const int hf   = lane >> 4;
  const int l15  = lane & 15;

  __shared__ unsigned short sK[32 * LDK];        // [key][feat] natural
  __shared__ unsigned short sV[D_HEAD * LDV];    // [feat][key] transposed
  __shared__ unsigned short sP[4 * 16 * LDP];    // per-wave P scratch

  const size_t rstride = 3 * D_MODEL;
  const unsigned short* qptr = qkv + (size_t)b * SEQ * rstride + h * D_HEAD;
  const unsigned short* kptr = qptr + D_MODEL;
  const unsigned short* vptr = qptr + 2 * D_MODEL;

  // Q fragments for this wave's 16 rows (features 0..63) held in registers
  FragU qf[2];
  {
    int qr = q0 + wave * 16 + l15;
    const unsigned short* p = qptr + (size_t)qr * rstride;
    #pragma unroll
    for (int f = 0; f < 2; ++f) {
      qf[f].q[0] = *(const u32x4*)(p + f * 32 + hf * 8);
      qf[f].q[1] = *(const u32x4*)(p + f * 32 + hf * 8 + 16);
    }
  }

  const v8f vzero = {0.f,0.f,0.f,0.f,0.f,0.f,0.f,0.f};
  v8f o[4];
  for (int t = 0; t < 4; ++t) o[t] = vzero;
  float m_run[8], l_run[8];
  for (int r = 0; r < 8; ++r) { m_run[r] = -1e30f; l_run[r] = 0.f; }

  unsigned short* pw = sP + wave * 16 * LDP;
  const int kv_end = q0 + 64;

  for (int kv0 = 0; kv0 < kv_end; kv0 += 32) {
    #pragma unroll
    for (int c = 0; c < 2; ++c) {                 // K block via async DMA (natural layout)
      int chunk = tid + c * 128;
      int key = chunk >> 3;
      int f8  = (chunk & 7) * 8;
      ld128_to_lds(kptr + (size_t)(kv0 + key) * rstride + f8, sK + key * LDK + f8);
    }
    // V block: issue both global loads first, then scatter-transpose into LDS
    u32x4 vd[2];
    #pragma unroll
    for (int c = 0; c < 2; ++c) {
      int chunk = tid + c * 128;
      int key = chunk >> 3;
      int f8  = (chunk & 7) * 8;
      vd[c] = *(const u32x4*)(vptr + (size_t)(kv0 + key) * rstride + f8);
    }
    #pragma unroll
    for (int c = 0; c < 2; ++c) {
      int chunk = tid + c * 128;
      int key = chunk >> 3;
      int f8  = (chunk & 7) * 8;
      unsigned short tmp[8];
      *(u32x4*)tmp = vd[c];
      #pragma unroll
      for (int j = 0; j < 8; ++j) sV[(f8 + j) * LDV + key] = tmp[j];
    }
    wait_async_le0();
    __syncthreads();

    // S = Q K^T (16q x 32keys): two 16-key subtiles, contraction over 64 feats
    v8f s[2];
    #pragma unroll
    for (int kt = 0; kt < 2; ++kt) {
      v8f sc = vzero;
      #pragma unroll
      for (int f = 0; f < 2; ++f) {
        FragU bfr;
        const unsigned short* p = sK + (kt * 16 + l15) * LDK + f * 32 + hf * 8;
        bfr.q[0] = *(const u32x4*)p;
        bfr.q[1] = *(const u32x4*)(p + 16);
        sc = __builtin_amdgcn_wmma_f32_16x16x32_bf16(false, qf[f].v, false, bfr.v,
                                                     (short)0, sc, false, false);
      }
      s[kt] = sc;
    }

    // scale, causal mask, online softmax (rows live in vgpr index, cols across 16 lanes)
    float alpha[8];
    #pragma unroll
    for (int r = 0; r < 8; ++r) {
      int qr = q0 + wave * 16 + hf * 8 + r;
      float mx = -1e30f;
      #pragma unroll
      for (int kt = 0; kt < 2; ++kt) {
        float val = s[kt][r] * 0.125f;            // 1/sqrt(64)
        int key = kv0 + kt * 16 + l15;
        if (key > qr) val = -1e30f;
        s[kt][r] = val;
        mx = fmaxf(mx, val);
      }
      mx = red16_max(mx);
      float mnew = fmaxf(m_run[r], mx);
      alpha[r] = __expf(m_run[r] - mnew);
      m_run[r] = mnew;
    }
    #pragma unroll
    for (int r = 0; r < 8; ++r) {
      float rowsum = 0.f;
      #pragma unroll
      for (int kt = 0; kt < 2; ++kt) {
        float p = __expf(s[kt][r] - m_run[r]);
        rowsum += p;
        pw[(hf * 8 + r) * LDP + kt * 16 + l15] = f2bf(p);
      }
      rowsum = red16_sum(rowsum);
      l_run[r] = l_run[r] * alpha[r] + rowsum;
    }
    #pragma unroll
    for (int t = 0; t < 4; ++t)
      #pragma unroll
      for (int r = 0; r < 8; ++r)
        o[t][r] *= alpha[r];

    // O += P(16x32) x V(32x64): P A-fragment via per-wave LDS (same-wave DS is in-order)
    FragU pf;
    pf.q[0] = *(const u32x4*)(pw + l15 * LDP + hf * 8);
    pf.q[1] = *(const u32x4*)(pw + l15 * LDP + hf * 8 + 16);
    #pragma unroll
    for (int t = 0; t < 4; ++t) {
      FragU vf;
      const unsigned short* p = sV + (t * 16 + l15) * LDV + hf * 8;
      vf.q[0] = *(const u32x4*)p;
      vf.q[1] = *(const u32x4*)(p + 16);
      o[t] = __builtin_amdgcn_wmma_f32_16x16x32_bf16(false, pf.v, false, vf.v,
                                                     (short)0, o[t], false, false);
    }
    __syncthreads();
  }

  #pragma unroll
  for (int t = 0; t < 4; ++t) {
    #pragma unroll
    for (int r = 0; r < 8; ++r) {
      int qr = q0 + wave * 16 + hf * 8 + r;
      float val = o[t][r] / l_run[r];
      outp[((size_t)b * SEQ + qr) * D_MODEL + h * D_HEAD + t * 16 + l15] = f2bf(val);
    }
  }
}

// ---------------------------------------------------------------- launch
extern "C" void kernel_launch(void* const* d_in, const int* in_sizes, int n_in,
                              void* d_out, int out_size, void* d_ws, size_t ws_size,
                              hipStream_t stream) {
  (void)in_sizes; (void)n_in; (void)out_size; (void)ws_size;
  const float* x     = (const float*)d_in[0];
  const float* w_qkv = (const float*)d_in[1];
  const float* w_out = (const float*)d_in[2];
  const float* ln1_g = (const float*)d_in[3];
  const float* ln1_b = (const float*)d_in[4];
  const float* ln2_g = (const float*)d_in[5];
  const float* ln2_b = (const float*)d_in[6];
  const float* w1    = (const float*)d_in[7];
  const float* b1    = (const float*)d_in[8];
  const float* w2    = (const float*)d_in[9];
  const float* b2    = (const float*)d_in[10];
  float* out = (float*)d_out;

  // scratch layout (64 MB, lifetime-based reuse)
  char* ws = (char*)d_ws;
  unsigned short* act  = (unsigned short*)(ws);                  //  8 MB: ln1 / attn-out / ln2
  unsigned short* wbuf = (unsigned short*)(ws + (8u  << 20));    //  8 MB: bf16 transposed weights (JIT)
  unsigned short* big  = (unsigned short*)(ws + (16u << 20));    // 32 MB: qkv (24MB) then gelu (32MB)
  float*          x2   = (float*)         (ws + (48u << 20));    // 16 MB: residual after attention

  // 1) LN1 -> bf16
  layernorm_to_bf16<<<BT, 256, 0, stream>>>(x, ln1_g, ln1_b, act);
  // 2) qkv = ln1 @ w_qkv  (bf16 out)
  cvt_transpose_bf16<<<dim3(3 * D_MODEL / 32, D_MODEL / 32), 256, 0, stream>>>(w_qkv, wbuf, D_MODEL, 3 * D_MODEL);
  gemm_bf16_wmma<0><<<dim3(3 * D_MODEL / BNT, BT / BMT), 256, 0, stream>>>(
      act, wbuf, big, nullptr, nullptr, BT, 3 * D_MODEL, D_MODEL);
  // 3) attention -> act (bf16, [B,T,C])
  attn_wmma<<<dim3(SEQ / 64, BATCH * N_HEADS), 128, 0, stream>>>(big, act);
  // 4) x2 = x + attn @ w_out
  cvt_transpose_bf16<<<dim3(D_MODEL / 32, D_MODEL / 32), 256, 0, stream>>>(w_out, wbuf, D_MODEL, D_MODEL);
  gemm_bf16_wmma<2><<<dim3(D_MODEL / BNT, BT / BMT), 256, 0, stream>>>(
      act, wbuf, x2, nullptr, x, BT, D_MODEL, D_MODEL);
  // 5) LN2 -> bf16
  layernorm_to_bf16<<<BT, 256, 0, stream>>>(x2, ln2_g, ln2_b, act);
  // 6) big = gelu(ln2 @ w1 + b1)  (bf16 out)
  cvt_transpose_bf16<<<dim3(D_FF / 32, D_MODEL / 32), 256, 0, stream>>>(w1, wbuf, D_MODEL, D_FF);
  gemm_bf16_wmma<1><<<dim3(D_FF / BNT, BT / BMT), 256, 0, stream>>>(
      act, wbuf, big, b1, nullptr, BT, D_FF, D_MODEL);
  // 7) out = x2 + big @ w2 + b2
  cvt_transpose_bf16<<<dim3(D_MODEL / 32, D_FF / 32), 256, 0, stream>>>(w2, wbuf, D_FF, D_MODEL);
  gemm_bf16_wmma<3><<<dim3(D_MODEL / BNT, BT / BMT), 256, 0, stream>>>(
      big, wbuf, (void*)out, b2, x2, BT, D_MODEL, D_FF);
}
